// Decoder_74998718923424
// MI455X (gfx1250) — matrix operand order repaired and checked by reading, not compiled, not run on hardware
//
#include <hip/hip_runtime.h>
#include <hip/hip_bf16.h>

typedef __attribute__((ext_vector_type(16))) _Float16 v16h;
typedef __attribute__((ext_vector_type(8)))  float    v8f;
typedef __attribute__((ext_vector_type(4)))  unsigned int u32x4;
typedef __attribute__((ext_vector_type(8)))  int          i32x8;
typedef __attribute__((ext_vector_type(4)))  int          i32x4;

#define B_  2
#define T_  1024
#define D_  512
#define H_  8
#define HD_ 64
#define F_  2048
#define L_  2
#define BT_ (B_ * T_)

// Tensor Data Mover availability (device pass only; fallback = manual staging)
#if defined(__HIP_DEVICE_COMPILE__) && defined(__gfx1250__) && \
    __has_builtin(__builtin_amdgcn_tensor_load_to_lds)
#define HAVE_TDM 1
#else
#define HAVE_TDM 0
#endif

// ---------------------------------------------------------------------------
// TDM helpers: 2D tile (tile_w x tile_h elements of f32, row stride `stride`
// elements) from global -> LDS, with 1-DWORD padding inserted after every
// 2^pad_code * 2 DWORDs (set pad_code so padding lands at each row end).
// D# layout per CDNA5 ISA.
// ---------------------------------------------------------------------------
__device__ __forceinline__ void tdm_load_2d(unsigned lds_off, const float* gptr,
                                            unsigned tile_w, unsigned tile_h,
                                            unsigned stride, unsigned pad_code) {
#if HAVE_TDM
  unsigned long long ga = (unsigned long long)(uintptr_t)gptr;
  u32x4 g0;
  g0.x = 1u;                                              // count=1, user desc
  g0.y = lds_off;                                         // lds_addr (bytes)
  g0.z = (unsigned)ga;                                    // global_addr[31:0]
  g0.w = (unsigned)((ga >> 32) & 0x01FFFFFFull) | 0x80000000u; // ga[56:32]|type=2
  i32x8 g1;
  g1[0] = (int)((2u << 16) | (1u << 20) | (pad_code << 22)); // 4B elem, pad_en,
                                                             // interval, amt=1DW
  g1[1] = 0;                                     // abar=0 | tdim0[15:0] (1M->0)
  g1[2] = (int)0x10u;                            // tdim0[31:16]=16 | tdim1 lo=0
  g1[3] = (int)(0x10u | (tile_w << 16));         // tdim1[31:16]=16 | tile_dim0
  g1[4] = (int)tile_h;                           // tile_dim1 | tile_dim2=0
  g1[5] = (int)stride;                           // tensor_dim0_stride[31:0]
  g1[6] = 0;
  g1[7] = 0;
  i32x4 z4 = (i32x4)0;
#if __clang_major__ >= 23
  i32x8 z8 = (i32x8)0;
  __builtin_amdgcn_tensor_load_to_lds(g0, g1, z4, z4, z8, 0);
#else
  __builtin_amdgcn_tensor_load_to_lds(g0, g1, z4, z4, 0);
#endif
#endif
}

__device__ __forceinline__ void tdm_wait0() {
#if __has_builtin(__builtin_amdgcn_s_wait_tensorcnt)
  __builtin_amdgcn_s_wait_tensorcnt((short)0);
#elif defined(__HIP_DEVICE_COMPILE__) && defined(__gfx1250__)
  asm volatile("s_wait_tensorcnt 0x0" ::: "memory");
#endif
}

// ---------------------------------------------------------------------------
// Embedding + sinusoidal positional encoding
// ---------------------------------------------------------------------------
__global__ __launch_bounds__(256) void embed_kernel(const int* __restrict__ tok,
                                                    const float* __restrict__ emb,
                                                    float* __restrict__ y) {
  int row = blockIdx.x;            // 0 .. BT-1
  int t   = row % T_;
  const float* e = emb + (size_t)tok[row] * D_;
  for (int d = threadIdx.x; d < D_; d += 256) {
    float i2    = (float)((d >> 1) << 1);
    float denom = __powf(10000.0f, i2 / (float)D_);
    float ang   = (float)t / denom;
    float pe    = (d & 1) ? __cosf(ang) : __sinf(ang);
    y[(size_t)row * D_ + d] = e[d] + pe;
  }
}

// ---------------------------------------------------------------------------
// Generic GEMM:  C[M,N] = act( A[M,K] @ W[K,N] + bias )   (f16 WMMA, f32 acc)
// Block: 256 threads (8 waves). Tile: 128(M) x 64(N), K-step 32.
// Tiles staged to LDS as f32 by the Tensor Data Mover (wave 0 issues the DMA,
// hardware inserts the +1 DWORD/row bank padding); f16 convert at frag read.
// ---------------------------------------------------------------------------
#define GBM 128
#define GBN 64
#define GBK 32

__global__ __launch_bounds__(256) void gemm_kernel(
    const float* __restrict__ A, int lda,
    const float* __restrict__ W, int ldw,
    const float* __restrict__ bias,
    float* __restrict__ C, int ldc,
    int K, int relu)
{
  __shared__ float As[GBM][GBK + 1];   // A tile f32, row-major, +1 pad
  __shared__ float Bs[GBK][GBN + 1];   // B tile f32, [k][n], +1 pad

  const int m0   = blockIdx.y * GBM;
  const int n0   = blockIdx.x * GBN;
  const int tid  = threadIdx.x;
  const int wave = tid >> 5;
  const int lane = tid & 31;
  const int lh   = lane >> 4;      // half-wave select (K range / row group)
  const int lm   = lane & 15;

#if HAVE_TDM
  const unsigned ldsA = (unsigned)(uintptr_t)&As[0][0];
  const unsigned ldsB = (unsigned)(uintptr_t)&Bs[0][0];
#endif

  v8f acc[4] = {};

  for (int k0 = 0; k0 < K; k0 += GBK) {
    // prefetch next K tile into L2 (speculative, gfx1250 global_prefetch)
    if (k0 + GBK < K) {
      __builtin_prefetch(&A[(size_t)(m0 + (tid >> 1)) * lda + k0 + GBK], 0, 0);
      __builtin_prefetch(&W[(size_t)(k0 + GBK + (tid >> 3)) * ldw + n0], 0, 0);
    }
#if HAVE_TDM
    if (wave == 0) {
      // A: 32 x 128 tile, pad after every 32 DWORDs (code 4)
      tdm_load_2d(ldsA, A + (size_t)m0 * lda + k0, GBK, GBM, (unsigned)lda, 4u);
      // B: 64 x 32 tile, pad after every 64 DWORDs (code 5)
      tdm_load_2d(ldsB, W + (size_t)k0 * ldw + n0, GBN, GBK, (unsigned)ldw, 5u);
      tdm_wait0();
    }
#else
    for (int i = tid; i < GBM * GBK; i += 256) {
      int r = i >> 5, c = i & 31;
      As[r][c] = A[(size_t)(m0 + r) * lda + k0 + c];
    }
    for (int i = tid; i < GBK * GBN; i += 256) {
      int kk = i >> 6, n = i & 63;
      Bs[kk][n] = W[(size_t)(k0 + kk) * ldw + n0 + n];
    }
#endif
    __syncthreads();

    // A fragment (16x32): lane = M row, half selects K range
    v16h a;
    const int arow = wave * 16 + lm;
#pragma unroll
    for (int i = 0; i < 8; i++) {
      int kk       = ((i < 4) ? 0 : 16) + lh * 8 + (i & 3) * 2;
      a[2 * i]     = (_Float16)As[arow][kk];
      a[2 * i + 1] = (_Float16)As[arow][kk + 1];
    }
#pragma unroll
    for (int g = 0; g < 4; g++) {
      // B fragment (32x16): lane%16 = N, lanes 0-15 K=0-7,16-23, 16-31 K=8-15,24-31
      v16h bf;
      const int bn = g * 16 + lm;
#pragma unroll
      for (int hh = 0; hh < 16; hh++) bf[hh] = (_Float16)Bs[((hh >> 3) << 4) + lh * 8 + (hh & 7)][bn];
      acc[g] = __builtin_amdgcn_wmma_f32_16x16x32_f16(
          false, a, false, bf, (short)0, acc[g], false, false);
    }
    __syncthreads();
  }

  // Write-out: lane holds column n, rows (lh*8 + r)
#pragma unroll
  for (int g = 0; g < 4; g++) {
    const int n  = n0 + g * 16 + lm;
    const float bv = bias[n];
#pragma unroll
    for (int r = 0; r < 8; r++) {
      const int m = m0 + wave * 16 + lh * 8 + r;
      float v = acc[g][r] + bv;
      if (relu) v = v > 0.0f ? v : 0.0f;
      C[(size_t)m * ldc + n] = v;
    }
  }
}

// ---------------------------------------------------------------------------
// Flash attention: one wave per (b, h, 16-row Q tile). Online softmax over
// T keys in chunks of 32. S = Q.K^T (4 WMMA), P.V (4 WMMA) per chunk.
// ---------------------------------------------------------------------------
__global__ __launch_bounds__(32) void attn_kernel(
    const float* __restrict__ Q, int ldq,
    const float* __restrict__ Kp, int ldk,
    const float* __restrict__ Vp, int ldv,
    const float* __restrict__ mask,
    float* __restrict__ O)
{
  const int qt   = blockIdx.x;     // 0 .. T/16-1
  const int h    = blockIdx.y;
  const int b    = blockIdx.z;
  const int lane = threadIdx.x;
  const int lh   = lane >> 4;
  const int lm   = lane & 15;

  __shared__ float    Sbuf[16][33];
  __shared__ _Float16 Pbuf[16][34];
  __shared__ float    alphaS[16];
  __shared__ float    lS[16];

  // Q fragments: 16x64 split into two 16x32 A-fragments
  v16h qa[2];
  const float* qbase = Q + ((size_t)(b * T_) + qt * 16 + lm) * ldq + h * HD_;
#pragma unroll
  for (int f = 0; f < 2; f++)
#pragma unroll
    for (int i = 0; i < 8; i++) {
      int kk = f * 32 + ((i < 4) ? 0 : 16) + lh * 8 + (i & 3) * 2;
      qa[f][2 * i]     = (_Float16)qbase[kk];
      qa[f][2 * i + 1] = (_Float16)qbase[kk + 1];
    }

  v8f oacc[4] = {};
  float m_run = -1e30f;            // valid in lanes 0-15 (row = lane)
  float l_run = 0.0f;
  const float scale = 0.125f;      // 1/sqrt(64)

  for (int kc = 0; kc < T_; kc += 32) {
    // ---- S = Q . K^T for 32 keys: two 16x16 C tiles, K-dim 64 (2 steps) ----
    v8f s[2] = {};
#pragma unroll
    for (int j = 0; j < 2; j++) {
      const int key = kc + j * 16 + lm;
      const float* kb0 = Kp + ((size_t)(b * T_) + key) * ldk + h * HD_;
#pragma unroll
      for (int f = 0; f < 2; f++) {
        v16h kb;
        const float* kbase = kb0 + f * 32 + lh * 8;
#pragma unroll
        for (int hh = 0; hh < 16; hh++) kb[hh] = (_Float16)kbase[((hh >> 3) << 4) + (hh & 7)];
        s[j] = __builtin_amdgcn_wmma_f32_16x16x32_f16(
            false, qa[f], false, kb, (short)0, s[j], false, false);
      }
    }
    // ---- scale + additive mask, stage to LDS ----
#pragma unroll
    for (int j = 0; j < 2; j++) {
      const int key = kc + j * 16 + lm;
#pragma unroll
      for (int r = 0; r < 8; r++) {
        const int row  = lh * 8 + r;
        const int qrow = qt * 16 + row;
        const float mv = mask[((size_t)(b * T_) + qrow) * T_ + key];
        Sbuf[row][j * 16 + lm] = s[j][r] * scale + mv;
      }
    }
    __syncthreads();
    // ---- online softmax (row stats in lanes 0-15) ----
    if (lane < 16) {
      const int row = lane;
      float rowmax = -1e30f;
      for (int c = 0; c < 32; c++) rowmax = fmaxf(rowmax, Sbuf[row][c]);
      float m_new = fmaxf(m_run, rowmax);
      float alpha = __expf(m_run - m_new);
      float sum = 0.0f;
      for (int c = 0; c < 32; c++) {
        float p = __expf(Sbuf[row][c] - m_new);
        Pbuf[row][c] = (_Float16)p;
        sum += p;
      }
      l_run = l_run * alpha + sum;
      m_run = m_new;
      alphaS[row] = alpha;
    }
    __syncthreads();
    // ---- rescale O accumulators ----
    float ar[8];
#pragma unroll
    for (int r = 0; r < 8; r++) ar[r] = alphaS[lh * 8 + r];
#pragma unroll
    for (int g = 0; g < 4; g++)
#pragma unroll
      for (int r = 0; r < 8; r++) oacc[g][r] *= ar[r];
    // ---- P A-fragment (16x32) ----
    v16h pa;
#pragma unroll
    for (int i = 0; i < 8; i++) {
      int kk = ((i < 4) ? 0 : 16) + lh * 8 + (i & 3) * 2;
      pa[2 * i]     = Pbuf[lm][kk];
      pa[2 * i + 1] = Pbuf[lm][kk + 1];
    }
    // ---- O += P . V  (4 WMMA over d groups) ----
#pragma unroll
    for (int g = 0; g < 4; g++) {
      v16h vb;
      const float* vbase =
          Vp + ((size_t)(b * T_) + kc + lh * 8) * ldv + h * HD_ + g * 16 + lm;
#pragma unroll
      for (int hh = 0; hh < 16; hh++) vb[hh] = (_Float16)vbase[(size_t)(((hh >> 3) << 4) + (hh & 7)) * ldv];
      oacc[g] = __builtin_amdgcn_wmma_f32_16x16x32_f16(
          false, pa, false, vb, (short)0, oacc[g], false, false);
    }
    __syncthreads();
  }

  if (lane < 16) lS[lane] = l_run;
  __syncthreads();
  float linv[8];
#pragma unroll
  for (int r = 0; r < 8; r++) linv[r] = 1.0f / lS[lh * 8 + r];
#pragma unroll
  for (int g = 0; g < 4; g++)
#pragma unroll
    for (int r = 0; r < 8; r++) {
      const int m = qt * 16 + lh * 8 + r;
      O[((size_t)(b * T_) + m) * D_ + h * HD_ + g * 16 + lm] = oacc[g][r] * linv[r];
    }
}

// ---------------------------------------------------------------------------
// Fused residual + LayerNorm: out = LN(a + r) * g + be   (one row per block)
// ---------------------------------------------------------------------------
__global__ __launch_bounds__(256) void ln_kernel(
    const float* __restrict__ a, const float* __restrict__ r,
    const float* __restrict__ g, const float* __restrict__ be,
    float* __restrict__ out)
{
  __shared__ float red[256];
  const int row = blockIdx.x;
  const int tid = threadIdx.x;
  float v[2];
  float s = 0.0f;
#pragma unroll
  for (int i = 0; i < 2; i++) {
    const int d = tid + i * 256;
    v[i] = a[(size_t)row * D_ + d] + r[(size_t)row * D_ + d];
    s += v[i];
  }
  red[tid] = s;
  __syncthreads();
  for (int off = 128; off > 0; off >>= 1) {
    if (tid < off) red[tid] += red[tid + off];
    __syncthreads();
  }
  const float mean = red[0] / (float)D_;
  __syncthreads();
  float s2 = 0.0f;
#pragma unroll
  for (int i = 0; i < 2; i++) { float d = v[i] - mean; s2 += d * d; }
  red[tid] = s2;
  __syncthreads();
  for (int off = 128; off > 0; off >>= 1) {
    if (tid < off) red[tid] += red[tid + off];
    __syncthreads();
  }
  const float rstd = rsqrtf(red[0] / (float)D_ + 1e-5f);
#pragma unroll
  for (int i = 0; i < 2; i++) {
    const int d = tid + i * 256;
    out[(size_t)row * D_ + d] = (v[i] - mean) * rstd * g[d] + be[d];
  }
}

// ---------------------------------------------------------------------------
extern "C" void kernel_launch(void* const* d_in, const int* in_sizes, int n_in,
                              void* d_out, int out_size, void* d_ws, size_t ws_size,
                              hipStream_t stream) {
  const float* x       = (const float*)d_in[0];
  const int*   tok     = (const int*)  d_in[1];
  const float* smask   = (const float*)d_in[2];
  const float* cmask   = (const float*)d_in[3];
  const float* emb     = (const float*)d_in[4];
  const float* Wqkv    = (const float*)d_in[5];
  const float* bqkv    = (const float*)d_in[6];
  const float* Wo      = (const float*)d_in[7];
  const float* bo      = (const float*)d_in[8];
  const float* Wkv     = (const float*)d_in[9];
  const float* bkv     = (const float*)d_in[10];
  const float* Wq      = (const float*)d_in[11];
  const float* bq      = (const float*)d_in[12];
  const float* Wco     = (const float*)d_in[13];
  const float* bco     = (const float*)d_in[14];
  const float* W1      = (const float*)d_in[15];
  const float* b1      = (const float*)d_in[16];
  const float* W2      = (const float*)d_in[17];
  const float* b2      = (const float*)d_in[18];
  const float* g1      = (const float*)d_in[19];
  const float* be1     = (const float*)d_in[20];
  const float* g2      = (const float*)d_in[21];
  const float* be2     = (const float*)d_in[22];
  const float* g3      = (const float*)d_in[23];
  const float* be3     = (const float*)d_in[24];

  // workspace layout (floats)
  float* ws   = (float*)d_ws;
  float* y    = ws;                          // BT*D
  float* qkv  = y    + (size_t)BT_ * D_;     // BT*3D
  float* attn = qkv  + (size_t)BT_ * 3 * D_; // BT*D
  float* proj = attn + (size_t)BT_ * D_;     // BT*D
  float* kv   = proj + (size_t)BT_ * D_;     // BT*2D
  float* qb   = kv   + (size_t)BT_ * 2 * D_; // BT*D
  float* ff1  = qb   + (size_t)BT_ * D_;     // BT*F

  const dim3 blk256(256), blk32(32);
  const dim3 g_attn(T_ / 16, H_, B_);

  embed_kernel<<<BT_, blk256, 0, stream>>>(tok, emb, y);

  for (int l = 0; l < L_; l++) {
    const float* Wqkv_l = Wqkv + (size_t)l * D_ * 3 * D_;
    const float* Wo_l   = Wo   + (size_t)l * D_ * D_;
    const float* Wkv_l  = Wkv  + (size_t)l * D_ * 2 * D_;
    const float* Wq_l   = Wq   + (size_t)l * D_ * D_;
    const float* Wco_l  = Wco  + (size_t)l * D_ * D_;
    const float* W1_l   = W1   + (size_t)l * D_ * F_;
    const float* W2_l   = W2   + (size_t)l * F_ * D_;

    // ---- self-attention block ----
    gemm_kernel<<<dim3(3 * D_ / GBN, BT_ / GBM), blk256, 0, stream>>>(
        y, D_, Wqkv_l, 3 * D_, bqkv + (size_t)l * 3 * D_, qkv, 3 * D_, D_, 0);
    attn_kernel<<<g_attn, blk32, 0, stream>>>(
        qkv, 3 * D_, qkv + D_, 3 * D_, qkv + 2 * D_, 3 * D_, smask, attn);
    gemm_kernel<<<dim3(D_ / GBN, BT_ / GBM), blk256, 0, stream>>>(
        attn, D_, Wo_l, D_, bo + (size_t)l * D_, proj, D_, D_, 0);
    ln_kernel<<<BT_, blk256, 0, stream>>>(
        proj, y, g1 + (size_t)l * D_, be1 + (size_t)l * D_, y);

    // ---- cross-attention block ----
    gemm_kernel<<<dim3(2 * D_ / GBN, BT_ / GBM), blk256, 0, stream>>>(
        x, D_, Wkv_l, 2 * D_, bkv + (size_t)l * 2 * D_, kv, 2 * D_, D_, 0);
    gemm_kernel<<<dim3(D_ / GBN, BT_ / GBM), blk256, 0, stream>>>(
        y, D_, Wq_l, D_, bq + (size_t)l * D_, qb, D_, D_, 0);
    attn_kernel<<<g_attn, blk32, 0, stream>>>(
        qb, D_, kv, 2 * D_, kv + D_, 2 * D_, cmask, attn);
    gemm_kernel<<<dim3(D_ / GBN, BT_ / GBM), blk256, 0, stream>>>(
        attn, D_, Wco_l, D_, bco + (size_t)l * D_, proj, D_, D_, 0);
    ln_kernel<<<BT_, blk256, 0, stream>>>(
        proj, y, g2 + (size_t)l * D_, be2 + (size_t)l * D_, y);

    // ---- feed-forward block ----
    gemm_kernel<<<dim3(F_ / GBN, BT_ / GBM), blk256, 0, stream>>>(
        y, D_, W1_l, F_, b1 + (size_t)l * F_, ff1, F_, D_, 1);
    gemm_kernel<<<dim3(D_ / GBN, BT_ / GBM), blk256, 0, stream>>>(
        ff1, F_, W2_l, D_, b2 + (size_t)l * D_, proj, D_, F_, 0);
    ln_kernel<<<BT_, blk256, 0, stream>>>(
        proj, y, g3 + (size_t)l * D_, be3 + (size_t)l * D_, y);
  }

  hipMemcpyAsync(d_out, y, (size_t)BT_ * D_ * sizeof(float),
                 hipMemcpyDeviceToDevice, stream);
}
